// RNNDecoder_19920058319278
// MI455X (gfx1250) — matrix operand order repaired and checked by reading, not compiled, not run on hardware
//
#include <hip/hip_runtime.h>
#include <hip/hip_bf16.h>

// ---------------------------------------------------------------------------
// Types for CDNA5 WMMA (wave32, 16x16x32 bf16 -> f32) and TDM descriptors
// ---------------------------------------------------------------------------
typedef __bf16 bf16;
typedef __attribute__((ext_vector_type(16))) __bf16 v16bf;
typedef __attribute__((ext_vector_type(8)))  __bf16 v8bf;
typedef __attribute__((ext_vector_type(8)))  float  v8f;
typedef unsigned int __attribute__((ext_vector_type(4))) v4u;
typedef int          __attribute__((ext_vector_type(8))) v8i;
typedef int          __attribute__((ext_vector_type(4))) v4i;

// Problem constants (match reference)
#define TT 128
#define BB 32
#define EE 512
#define HH 512
#define IH 1024
#define VV 32000

// Projection GEMM tiling
#define PROJ_BM 64     // block rows
#define PROJ_BN 256    // block cols
#define PROJ_KC 64     // K chunk staged per TDM transfer
#define LDS_STRIDE 72  // 64 K elems + 8 pad elems (TDM pad: 4 dwords / 32 dwords)

// ---------------------------------------------------------------------------
// Fragment loaders per CDNA5 ISA 7.12.2 VGPR layouts (wave32)
// A (16x32, MxK): lane L holds row r=L%16.  L<16: K=0..7 & 16..23,
//                 L>=16: K=8..15 & 24..31
// B (32x16, KxN): lane L holds col n=L%16.  L<16: K=0..7 & 16..23, L>=16: K=8..15 & 24..31
// C/D (16x16 f32): lane L col n=L%16; VGPR i holds row (L<16 ? i : 8+i)
// ---------------------------------------------------------------------------
__device__ __forceinline__ v16bf pack16(v8bf lo, v8bf hi) {
    v16bf out;
#pragma unroll
    for (int i = 0; i < 8; ++i) { out[i] = lo[i]; out[i + 8] = hi[i]; }
    return out;
}

__device__ __forceinline__ v16bf load_a_frag(const bf16* __restrict__ A, int lda,
                                             int m0, int k, int lane) {
    int r  = lane & 15;
    int kb = (lane < 16) ? 0 : 8;
    const bf16* p = A + (long)(m0 + r) * lda + k + kb;
    return pack16(*(const v8bf*)p, *(const v8bf*)(p + 16));
}

__device__ __forceinline__ v16bf load_b_frag(const bf16* __restrict__ W, int ldw,
                                             int n0, int k, int lane) {
    int n  = lane & 15;
    int kb = (lane < 16) ? 0 : 8;
    const bf16* p = W + (long)(n0 + n) * ldw + k + kb;
    return pack16(*(const v8bf*)p, *(const v8bf*)(p + 16));
}

// LDS-resident variants (addresses are uniform-strided, ds_load_b128)
__device__ __forceinline__ v16bf lds_a_frag(const bf16* sA, int lda, int k, int lane) {
    int r  = lane & 15;
    int kb = (lane < 16) ? 0 : 8;
    const bf16* p = sA + r * lda + k + kb;
    return pack16(*(const v8bf*)p, *(const v8bf*)(p + 16));
}

__device__ __forceinline__ v16bf lds_b_frag(const bf16* sB, int ldb, int k, int lane) {
    int n  = lane & 15;
    int kb = (lane < 16) ? 0 : 8;
    const bf16* p = sB + n * ldb + k + kb;
    return pack16(*(const v8bf*)p, *(const v8bf*)(p + 16));
}

__device__ __forceinline__ v8f wmma_bf16(v16bf a, v16bf b, v8f c) {
    return __builtin_amdgcn_wmma_f32_16x16x32_bf16(false, a, false, b,
                                                   (short)0, c, false, false);
}

__device__ __forceinline__ float sigmoidf_(float x) {
    return 1.0f / (1.0f + __expf(-x));
}

// ---------------------------------------------------------------------------
// Tensor Data Mover: async 2D tile load Global -> LDS (bf16, padded rows).
// D# packed per CDNA5 ISA.  Pad: 4 dwords every
// 32 dwords => LDS row stride = 64 elems + 8 = 144B (conflict-free ds_b128).
// Tracked by TENSORcnt; wait with s_wait_tensorcnt.
// ---------------------------------------------------------------------------
__device__ __forceinline__ void tdm_load_2d_bf16(unsigned lds_addr,
                                                 const void* gaddr,
                                                 unsigned tensor_d0,   // K extent
                                                 unsigned tensor_d1,   // row extent
                                                 unsigned tile_d0,     // K tile
                                                 unsigned tile_d1,     // row tile
                                                 unsigned long stride0) // elems/row
{
    unsigned long ga = (unsigned long)gaddr;
    v4u g0;
    g0[0] = 1u;                                        // count=1, user load
    g0[1] = lds_addr;                                  // LDS byte address
    g0[2] = (unsigned)(ga & 0xffffffffu);              // global_addr[31:0]
    g0[3] = (unsigned)((ga >> 32) & 0x01ffffffu) | (2u << 30);  // [56:32] | type=2
    v8i g1;
    unsigned d0 = (1u << 16)            // data_size = 1 (2 bytes)
                | (1u << 20)            // pad_enable
                | (4u << 22)            // pad_interval: every 32 dwords
                | (3u << 25);           // pad_amount: 4 dwords
    g1[0] = (int)d0;
    g1[1] = (int)((tensor_d0 & 0xffffu) << 16);                         // dim0 lo16
    g1[2] = (int)((tensor_d0 >> 16) | ((tensor_d1 & 0xffffu) << 16));   // dim0 hi|dim1 lo
    g1[3] = (int)((tensor_d1 >> 16) | ((tile_d0 & 0xffffu) << 16));     // dim1 hi|tile0
    g1[4] = (int)(tile_d1 & 0xffffu);                                   // tile1, tile2=0
    g1[5] = (int)(unsigned)(stride0 & 0xffffffffu);                     // stride0 lo32
    g1[6] = (int)(unsigned)((stride0 >> 32) & 0xffffu);                 // stride0 hi16
    g1[7] = 0;
    v4i g2 = {0, 0, 0, 0};
    v4i g3 = {0, 0, 0, 0};
#if defined(__clang_major__) && (__clang_major__ >= 23)
    v8i g4 = {0, 0, 0, 0, 0, 0, 0, 0};
    __builtin_amdgcn_tensor_load_to_lds(g0, g1, g2, g3, g4, 0);
#else
    __builtin_amdgcn_tensor_load_to_lds(g0, g1, g2, g3, 0);
#endif
}

__device__ __forceinline__ unsigned lds_byte_off(const void* p) {
    return (unsigned)(unsigned long)(uintptr_t)p;
}

// ---------------------------------------------------------------------------
// 0) f32 -> bf16 bulk convert (grid-stride)
// ---------------------------------------------------------------------------
__global__ void f32_to_bf16_kernel(const float* __restrict__ src,
                                   bf16* __restrict__ dst, long n) {
    long i = (long)blockIdx.x * blockDim.x + threadIdx.x;
    long stride = (long)gridDim.x * blockDim.x;
    for (; i < n; i += stride) dst[i] = (bf16)src[i];
}

// ---------------------------------------------------------------------------
// 1) Bridge: h0 = tanh(init_hidden @ W_bridge^T + b_bridge)   [B, H]
// ---------------------------------------------------------------------------
__global__ void bridge_kernel(const float* __restrict__ init_hidden,  // [B, IH]
                              const float* __restrict__ W_bridge,     // [H, IH]
                              const float* __restrict__ b_bridge,     // [H]
                              float* __restrict__ c0,                 // [B, H]
                              bf16* __restrict__ h0_bf) {             // [B, H]
    int idx = blockIdx.x * blockDim.x + threadIdx.x;
    if (idx >= BB * HH) return;
    int b = idx / HH, j = idx % HH;
    const float* x = init_hidden + (long)b * IH;
    const float* w = W_bridge + (long)j * IH;
    float acc = 0.f;
#pragma unroll 4
    for (int k = 0; k < IH; ++k) acc += x[k] * w[k];
    float h = tanhf(acc + b_bridge[j]);
    c0[idx] = h;
    h0_bf[idx] = (bf16)h;
}

// ---------------------------------------------------------------------------
// 2) xW = trg_embed @ W_ih^T + (b_ih + b_hh)    [T*B=4096, 4H=2048], K=512
// ---------------------------------------------------------------------------
__global__ void __launch_bounds__(256)
xw_gemm_kernel(const bf16* __restrict__ A,    // [4096, 512]
               const bf16* __restrict__ W,    // [2048, 512]  (W_ih)
               const float* __restrict__ b_ih,
               const float* __restrict__ b_hh,
               float* __restrict__ C) {       // [4096, 2048]
    const int K = EE, ldc = 4 * HH;
    int lane = threadIdx.x & 31;
    int wid  = threadIdx.x >> 5;
    int m0 = blockIdx.y * 32 + (wid >> 2) * 16;
    int n0 = blockIdx.x * 128 + (wid & 3) * 32;
    v8f acc0 = {}, acc1 = {};
    for (int k = 0; k < K; k += 32) {
        v16bf a  = load_a_frag(A, K, m0, k, lane);
        v16bf b0 = load_b_frag(W, K, n0, k, lane);
        v16bf b1 = load_b_frag(W, K, n0 + 16, k, lane);
        acc0 = wmma_bf16(a, b0, acc0);
        acc1 = wmma_bf16(a, b1, acc1);
    }
    int n = lane & 15, mb = (lane < 16) ? 0 : 8;
    float bv0 = b_ih[n0 + n] + b_hh[n0 + n];
    float bv1 = b_ih[n0 + 16 + n] + b_hh[n0 + 16 + n];
#pragma unroll
    for (int i = 0; i < 8; ++i) {
        C[(long)(m0 + mb + i) * ldc + n0 + n]      = acc0[i] + bv0;
        C[(long)(m0 + mb + i) * ldc + n0 + 16 + n] = acc1[i] + bv1;
    }
}

// ---------------------------------------------------------------------------
// 3) One LSTM timestep: gates = xW[t] + h @ W_hh^T ; c,h update fused.
//    grid.x = H/16 = 32 column-groups; block = 128 thr = 4 waves (one/gate)
// ---------------------------------------------------------------------------
__global__ void __launch_bounds__(128)
lstm_step_kernel(const bf16* __restrict__ h_in,    // [B, H] bf16 (ping)
                 const bf16* __restrict__ W_hh,    // [4H, H] bf16
                 const float* __restrict__ xW_t,   // [B, 4H] (this timestep)
                 float* __restrict__ c_buf,        // [B, H] f32 state
                 float* __restrict__ out_f32_t,    // d_out outputs[t]: [B, H]
                 bf16*  __restrict__ h_out,        // [B, H] bf16 (pong)
                 bf16*  __restrict__ out_bf_t) {   // bf16 copy for projection
    __shared__ float g_lds[4][BB][16];
    const int K = HH;
    int lane = threadIdx.x & 31;
    int w = threadIdx.x >> 5;                  // gate: 0=i 1=f 2=g 3=o
    int j = blockIdx.x;                        // hidden column group
    int nW = w * HH + j * 16;                  // row offset into W_hh
    v8f acc0 = {}, acc1 = {};
    for (int k = 0; k < K; k += 32) {
        v16bf a0 = load_a_frag(h_in, K, 0, k, lane);   // batch rows 0..15
        v16bf a1 = load_a_frag(h_in, K, 16, k, lane);  // batch rows 16..31
        v16bf b  = load_b_frag(W_hh, K, nW, k, lane);
        acc0 = wmma_bf16(a0, b, acc0);
        acc1 = wmma_bf16(a1, b, acc1);
    }
    int n = lane & 15, mb = (lane < 16) ? 0 : 8;
#pragma unroll
    for (int i = 0; i < 8; ++i) {
        g_lds[w][mb + i][n]      = acc0[i];
        g_lds[w][16 + mb + i][n] = acc1[i];
    }
    __syncthreads();
    for (int e = threadIdx.x; e < BB * 16; e += 128) {
        int m = e >> 4, nn = e & 15;
        int hidx = j * 16 + nn;
        const float* xrow = xW_t + (long)m * (4 * HH);
        float ig = g_lds[0][m][nn] + xrow[hidx];
        float fg = g_lds[1][m][nn] + xrow[HH + hidx];
        float gg = g_lds[2][m][nn] + xrow[2 * HH + hidx];
        float og = g_lds[3][m][nn] + xrow[3 * HH + hidx];
        long sidx = (long)m * HH + hidx;
        float cn = sigmoidf_(fg) * c_buf[sidx] + sigmoidf_(ig) * tanhf(gg);
        float hn = sigmoidf_(og) * tanhf(cn);
        c_buf[sidx]     = cn;
        out_f32_t[sidx] = hn;
        h_out[sidx]     = (bf16)hn;
        out_bf_t[sidx]  = (bf16)hn;
    }
}

// ---------------------------------------------------------------------------
// 4) finalize hidden: hT = outputs[T-1], cT = c_buf
// ---------------------------------------------------------------------------
__global__ void finalize_kernel(const float* __restrict__ out_last,
                                const float* __restrict__ c_buf,
                                float* __restrict__ hT,
                                float* __restrict__ cT) {
    int i = blockIdx.x * blockDim.x + threadIdx.x;
    if (i >= BB * HH) return;
    hT[i] = out_last[i];
    cT[i] = c_buf[i];
}

// ---------------------------------------------------------------------------
// 5) projection: [4096,512] x [32000,512]^T + b_out -> [4096,32000]
//    TDM double-buffered LDS GEMM.  Block = 512 thr (16 waves, 4M x 4N),
//    block tile 64x256, K chunks of 64 staged by tensor_load_to_lds.
// ---------------------------------------------------------------------------
__global__ void __launch_bounds__(512)
proj_gemm_tdm_kernel(const bf16* __restrict__ A,     // [4096, 512]
                     const bf16* __restrict__ W,     // [32000, 512]
                     const float* __restrict__ bias, // [32000]
                     float* __restrict__ C) {        // [4096, 32000]
    __shared__ __align__(16) bf16 lsA[2][PROJ_BM * LDS_STRIDE];  // 2 x  9 KB
    __shared__ __align__(16) bf16 lsB[2][PROJ_BN * LDS_STRIDE];  // 2 x 36 KB

    int lane  = threadIdx.x & 31;
    int wid   = threadIdx.x >> 5;     // 0..15
    int waveM = wid >> 2;             // 0..3
    int waveN = wid & 3;              // 0..3
    long mB = (long)blockIdx.y * PROJ_BM;
    long nB = (long)blockIdx.x * PROJ_BN;

    // --- preload K-chunk 0 (wave 0 drives the TDM; TENSORcnt tracks) ---
    if (wid == 0) {
        tdm_load_2d_bf16(lds_byte_off(&lsA[0][0]), A + mB * EE,
                         EE, (unsigned)(TT * BB), PROJ_KC, PROJ_BM, EE);
        tdm_load_2d_bf16(lds_byte_off(&lsB[0][0]), W + nB * EE,
                         EE, (unsigned)VV, PROJ_KC, PROJ_BN, EE);
        __builtin_amdgcn_s_wait_tensorcnt(0);
    }
    __syncthreads();

    v8f acc[4] = {};
    int p = 0;
    for (int kc = 0; kc < EE; kc += PROJ_KC) {
        // kick off next chunk into the other buffer while we compute
        if (wid == 0 && kc + PROJ_KC < EE) {
            tdm_load_2d_bf16(lds_byte_off(&lsA[p ^ 1][0]),
                             A + mB * EE + kc + PROJ_KC,
                             EE, (unsigned)(TT * BB), PROJ_KC, PROJ_BM, EE);
            tdm_load_2d_bf16(lds_byte_off(&lsB[p ^ 1][0]),
                             W + nB * EE + kc + PROJ_KC,
                             EE, (unsigned)VV, PROJ_KC, PROJ_BN, EE);
        }
        const bf16* sA = &lsA[p][(waveM * 16) * LDS_STRIDE];
        const bf16* sB = &lsB[p][(waveN * 64) * LDS_STRIDE];
#pragma unroll
        for (int k = 0; k < PROJ_KC; k += 32) {
            v16bf a = lds_a_frag(sA, LDS_STRIDE, k, lane);
#pragma unroll
            for (int t = 0; t < 4; ++t) {
                v16bf b = lds_b_frag(sB + t * 16 * LDS_STRIDE, LDS_STRIDE, k, lane);
                acc[t] = wmma_bf16(a, b, acc[t]);
            }
        }
        if (wid == 0) __builtin_amdgcn_s_wait_tensorcnt(0);
        __syncthreads();
        p ^= 1;
    }

    // --- epilogue: bias fused store ---
    int n = lane & 15, mb = (lane < 16) ? 0 : 8;
    long mBase = mB + waveM * 16 + mb;
    long nBase = nB + waveN * 64;
#pragma unroll
    for (int t = 0; t < 4; ++t) {
        float bv = bias[nBase + t * 16 + n];
#pragma unroll
        for (int i = 0; i < 8; ++i)
            C[(mBase + i) * (long)VV + nBase + t * 16 + n] = acc[t][i] + bv;
    }
}

// ---------------------------------------------------------------------------
// Host launcher
// ---------------------------------------------------------------------------
extern "C" void kernel_launch(void* const* d_in, const int* in_sizes, int n_in,
                              void* d_out, int out_size, void* d_ws, size_t ws_size,
                              hipStream_t stream) {
    const float* trg_embed   = (const float*)d_in[0];  // [T,B,E]
    const float* init_hidden = (const float*)d_in[1];  // [B,IH]
    const float* W_ih        = (const float*)d_in[2];  // [4H,E]
    const float* W_hh        = (const float*)d_in[3];  // [4H,H]
    const float* b_ih        = (const float*)d_in[4];  // [4H]
    const float* b_hh        = (const float*)d_in[5];  // [4H]
    const float* W_bridge    = (const float*)d_in[6];  // [H,IH]
    const float* b_bridge    = (const float*)d_in[7];  // [H]
    const float* W_out       = (const float*)d_in[8];  // [V,H]
    const float* b_out       = (const float*)d_in[9];  // [V]
    float* out = (float*)d_out;

    // d_out layout: outputs [T,B,H] | hT [B,H] | cT [B,H] | projected [T,B,V]
    const long OUT_HT   = (long)TT * BB * HH;          // 2,097,152
    const long OUT_CT   = OUT_HT + (long)BB * HH;      // 2,113,536
    const long OUT_PROJ = OUT_CT + (long)BB * HH;      // 2,129,920

    // workspace layout (bytes)
    char* ws = (char*)d_ws;
    bf16* trg_bf  = (bf16*)(ws + 0);                    //  4,194,304
    bf16* Wih_bf  = (bf16*)(ws + 4194304);              //  2,097,152
    bf16* Whh_bf  = (bf16*)(ws + 6291456);              //  2,097,152
    bf16* Wout_bf = (bf16*)(ws + 8388608);              // 32,768,000
    float* xW     = (float*)(ws + 41156608);            // 33,554,432
    bf16* out_bf  = (bf16*)(ws + 74711040);             //  4,194,304
    float* c_buf  = (float*)(ws + 78905344);            //     65,536
    bf16* h_ping  = (bf16*)(ws + 78970880);             // 2 x 32,768

    // --- convert weights/activations to bf16 ---
    f32_to_bf16_kernel<<<1024, 256, 0, stream>>>(trg_embed, trg_bf, (long)TT * BB * EE);
    f32_to_bf16_kernel<<<1024, 256, 0, stream>>>(W_ih, Wih_bf, (long)4 * HH * EE);
    f32_to_bf16_kernel<<<1024, 256, 0, stream>>>(W_hh, Whh_bf, (long)4 * HH * HH);
    f32_to_bf16_kernel<<<2048, 256, 0, stream>>>(W_out, Wout_bf, (long)VV * HH);

    // --- bridge: h0 = c0 = tanh(init_hidden @ W_bridge^T + b_bridge) ---
    bridge_kernel<<<(BB * HH + 255) / 256, 256, 0, stream>>>(
        init_hidden, W_bridge, b_bridge, c_buf, h_ping);

    // --- hoisted input GEMM: xW = X @ W_ih^T + b_ih + b_hh ---
    {
        dim3 grid(4 * HH / 128, TT * BB / 32);  // (16, 128)
        xw_gemm_kernel<<<grid, 256, 0, stream>>>(trg_bf, Wih_bf, b_ih, b_hh, xW);
    }

    // --- sequential LSTM recurrence (128 small WMMA kernels, ping-pong h) ---
    for (int t = 0; t < TT; ++t) {
        const bf16* hin = h_ping + (long)(t & 1) * BB * HH;
        bf16* hout      = h_ping + (long)((t + 1) & 1) * BB * HH;
        lstm_step_kernel<<<HH / 16, 128, 0, stream>>>(
            hin, Whh_bf,
            xW + (long)t * BB * 4 * HH,
            c_buf,
            out + (long)t * BB * HH,
            hout,
            out_bf + (long)t * BB * HH);
    }

    // --- final hidden state ---
    finalize_kernel<<<(BB * HH + 255) / 256, 256, 0, stream>>>(
        out + (long)(TT - 1) * BB * HH, c_buf, out + OUT_HT, out + OUT_CT);

    // --- output projection: TDM double-buffered LDS WMMA GEMM ---
    {
        dim3 grid(VV / PROJ_BN, TT * BB / PROJ_BM);  // (125, 64)
        proj_gemm_tdm_kernel<<<grid, 512, 0, stream>>>(out_bf, Wout_bf, b_out,
                                                       out + OUT_PROJ);
    }
}
